// VariationalAttention_87471303950758
// MI455X (gfx1250) — matrix operand order repaired and checked by reading, not compiled, not run on hardware
//
#include <hip/hip_runtime.h>
#include <hip/hip_bf16.h>

// VariationalAttention for MI455X (gfx1250, wave32).
// All GEMMs on V_WMMA_F32_16X16X32_BF16 using split-precision bf16x3
// (x = hi + lo; accumulate hi*hi + hi*lo + lo*hi in f32): ~8x more K
// retired per matrix issue than f32-K4 WMMA, ~fp32 accuracy (2^-16 rel).
// Dominant (B*T*S)x100 . 100x100 GEMM fully fused: h1 hi/lo strips in LDS,
// h2 never materialized, mean/var reduced in-register + shfl_xor.
// Inner K-loops are guard-free (clamped B rows; garbage columns are finite
// and masked at the scalar epilogue), keeping the hot loop pure load+WMMA.

namespace {

constexpr int Bb = 16, Tt = 128, Ss = 128, Dd = 512, Hh = 100;
constexpr int HPL = 104;   // padded row stride (floats) for pt/ps
constexpr int KP  = 128;   // padded K for W2 (4 steps of 32)
constexpr int LSTR = 136;  // LDS strip row stride (bf16): 272B rows, conflict-free

typedef __attribute__((ext_vector_type(8)))  float  v8f;
typedef __attribute__((ext_vector_type(16))) __bf16 v16bf;
typedef unsigned short u16;
typedef unsigned int   u32;

union Frag16 { v16bf v; u32 u[8]; };

__device__ __forceinline__ v8f wmma_bf16(v16bf a, v16bf b, v8f c) {
  return __builtin_amdgcn_wmma_f32_16x16x32_bf16(false, a, false, b, (short)0, c,
                                                 false, false);
}

__device__ __forceinline__ float softplusf(float x) {
  return (x > 30.f) ? x : log1pf(expf(x));
}

__device__ __forceinline__ u16 f2bf(float f) {  // RNE
  u32 u = __float_as_uint(f);
  u += 0x7FFFu + ((u >> 16) & 1u);
  return (u16)(u >> 16);
}
__device__ __forceinline__ float bf2f(u16 h) {
  return __uint_as_float(((u32)h) << 16);
}

// A-fragment (16x32 bf16): row pre-offset by lane&15; elements 0..15 = K[k+2*ksA..+15],
// ksA = 8*(lane>=16).
__device__ __forceinline__ v16bf loadA16(const u16* row, int k, int ksA) {
  Frag16 f;
  const u32* p = (const u32*)(row + k + 2 * ksA);
#pragma unroll
  for (int j = 0; j < 8; ++j) f.u[j] = p[j];
  return f.v;
}

// B-fragment (32x16 bf16, W-form B[k][n]=W[n][k]): W row pre-offset by column;
// 16 contiguous K at k+ksB, ksB = 16*(lane>=16).
__device__ __forceinline__ v16bf loadB16(const u16* row, int k, int ksB) {
  Frag16 f;
  const u32* p = (const u32*)(row + k + ksB);
#pragma unroll
  for (int j = 0; j < 8; ++j) f.u[j] = p[j];
  return f.v;
}

// ---------------------------------------------------------------------------
// Elementwise hi/lo bf16 split.
// ---------------------------------------------------------------------------
__global__ void k_split(const float* __restrict__ src, u16* __restrict__ hi,
                        u16* __restrict__ lo, int n) {
  int i = blockIdx.x * blockDim.x + threadIdx.x;
  if (i >= n) return;
  float f = src[i];
  u16 h = f2bf(f);
  hi[i] = h;
  lo[i] = f2bf(f - bf2f(h));
}

// Transpose-split memory bank: memT[b][d][s] = mem[b][s][d] (hi/lo bf16).
__global__ void k_tsplit_mem(const float* __restrict__ mem,
                             u16* __restrict__ th, u16* __restrict__ tl) {
  int i = blockIdx.x * blockDim.x + threadIdx.x;
  if (i >= Bb * Dd * Ss) return;
  int b = i / (Dd * Ss);
  int r = i - b * (Dd * Ss);
  int d = r >> 7;            // / Ss
  int s = r & (Ss - 1);
  float f = mem[(size_t)(b * Ss + s) * Dd + d];
  u16 h = f2bf(f);
  th[i] = h;
  tl[i] = f2bf(f - bf2f(h));
}

// Split W2 into hi/lo bf16, K zero-padded to KP=128. Layout [g][k], ld=KP.
__global__ void k_prep_w2(const float* __restrict__ W2,
                          u16* __restrict__ w2h, u16* __restrict__ w2l) {
  int idx = blockIdx.x * blockDim.x + threadIdx.x;
  if (idx >= Hh * KP) return;
  int g = idx >> 7;
  int k = idx & (KP - 1);
  float f = (k < Hh) ? W2[(size_t)g * Hh + k] : 0.f;
  u16 h = f2bf(f);
  w2h[idx] = h;
  w2l[idx] = f2bf(f - bf2f(h));
}

// ---------------------------------------------------------------------------
// Generic batched bf16x3 GEMM: C[M,N] = A[M,K] . W[N,K]^T
// (K multiple of 32; all tiles fully valid -> guard-free inner loop)
// ---------------------------------------------------------------------------
__global__ void k_gemm_bf3(const u16* __restrict__ Ah, const u16* __restrict__ Al,
                           const u16* __restrict__ Wh, const u16* __restrict__ Wl,
                           float* __restrict__ C,
                           int tilesM, int tilesN, int K,
                           int lda, int ldw, int ldc,
                           long batA, long batW, long batC, int nBatch) {
  int lane = threadIdx.x & 31;
  int wave = blockIdx.x * (blockDim.x >> 5) + (threadIdx.x >> 5);
  int tpb = tilesM * tilesN;
  if (wave >= tpb * nBatch) return;
  int bi = wave / tpb;
  int rt = wave - bi * tpb;
  int tm = rt / tilesN;
  int tn = rt - tm * tilesN;
  int r15 = lane & 15;
  int ksA = (lane >> 4) << 3;
  int ksB = (lane >> 4) << 4;
  const u16* arh = Ah + (size_t)bi * batA + (size_t)(tm * 16 + r15) * lda;
  const u16* arl = Al + (size_t)bi * batA + (size_t)(tm * 16 + r15) * lda;
  const u16* wrh = Wh + (size_t)bi * batW + (size_t)(tn * 16 + r15) * ldw;
  const u16* wrl = Wl + (size_t)bi * batW + (size_t)(tn * 16 + r15) * ldw;
  v8f c = {0.f, 0.f, 0.f, 0.f, 0.f, 0.f, 0.f, 0.f};
#pragma unroll 2
  for (int k = 0; k < K; k += 32) {
    v16bf ah = loadA16(arh, k, ksA);
    v16bf al = loadA16(arl, k, ksA);
    v16bf bh = loadB16(wrh, k, ksB);
    v16bf bl = loadB16(wrl, k, ksB);
    c = wmma_bf16(ah, bh, c);
    c = wmma_bf16(ah, bl, c);
    c = wmma_bf16(al, bh, c);
  }
  float* Cb = C + (size_t)bi * batC + (size_t)(tm * 16) * ldc + tn * 16;
  int rbase = (lane >> 4) << 3;
#pragma unroll
  for (int r = 0; r < 8; ++r) Cb[(size_t)(rbase + r) * ldc + r15] = c[r];
}

// ---------------------------------------------------------------------------
// Row softmax over S=128: normalized in-place + (T,B,S)-transposed to d_out.
// ---------------------------------------------------------------------------
__global__ void k_softmax(float* __restrict__ alignbuf, float* __restrict__ out_align) {
  int lane = threadIdx.x & 31;
  int row = blockIdx.x * (blockDim.x >> 5) + (threadIdx.x >> 5);
  if (row >= Bb * Tt) return;
  int b = row >> 7, t = row & (Tt - 1);
  float* p = alignbuf + (size_t)row * Ss;
  float v[4];
  float mx = -3.4e38f;
#pragma unroll
  for (int j = 0; j < 4; ++j) { v[j] = p[lane + 32 * j]; mx = fmaxf(mx, v[j]); }
#pragma unroll
  for (int m = 16; m >= 1; m >>= 1) mx = fmaxf(mx, __shfl_xor(mx, m, 32));
  float sum = 0.f;
#pragma unroll
  for (int j = 0; j < 4; ++j) { v[j] = expf(v[j] - mx); sum += v[j]; }
#pragma unroll
  for (int m = 16; m >= 1; m >>= 1) sum += __shfl_xor(sum, m, 32);
  float inv = 1.f / sum;
  float* po = out_align + ((size_t)t * Bb + b) * Ss;
#pragma unroll
  for (int j = 0; j < 4; ++j) {
    float av = v[j] * inv;
    p[lane + 32 * j] = av;
    po[lane + 32 * j] = av;
  }
}

// ---------------------------------------------------------------------------
// pt[b,t,h] = input . W1[h,:512] ; ps[b,s,h] = memory . W1[h,512:]  (bf16x3)
// B rows clamped for h>=100; garbage columns are finite and never stored.
// ---------------------------------------------------------------------------
__global__ void k_pt_ps_bf3(const u16* __restrict__ inH, const u16* __restrict__ inL,
                            const u16* __restrict__ memH, const u16* __restrict__ memL,
                            const u16* __restrict__ w1H, const u16* __restrict__ w1L,
                            float* __restrict__ ptb, float* __restrict__ psb) {
  const int tilesM = (Bb * Tt) / 16;  // 128
  const int tilesN = 7;
  int lane = threadIdx.x & 31;
  int wave = blockIdx.x * (blockDim.x >> 5) + (threadIdx.x >> 5);
  if (wave >= 2 * tilesM * tilesN) return;
  int job = wave / (tilesM * tilesN);
  int rt = wave - job * (tilesM * tilesN);
  int tm = rt / tilesN;
  int tn = rt - tm * tilesN;
  int r15 = lane & 15;
  int ksA = (lane >> 4) << 3;
  int ksB = (lane >> 4) << 4;
  const u16* arh = (job == 0 ? inH : memH) + (size_t)(tm * 16 + r15) * Dd;
  const u16* arl = (job == 0 ? inL : memL) + (size_t)(tm * 16 + r15) * Dd;
  int hrow = tn * 16 + r15;
  int hcl  = (hrow < Hh) ? hrow : 0;  // clamp: invalid columns masked at store
  int woff = (job == 0 ? 0 : Dd);
  const u16* wrh = w1H + woff + (size_t)hcl * (2 * Dd);
  const u16* wrl = w1L + woff + (size_t)hcl * (2 * Dd);
  float* C = (job == 0 ? ptb : psb);
  v8f c = {0.f, 0.f, 0.f, 0.f, 0.f, 0.f, 0.f, 0.f};
#pragma unroll 2
  for (int k = 0; k < Dd; k += 32) {
    v16bf ah = loadA16(arh, k, ksA);
    v16bf al = loadA16(arl, k, ksA);
    v16bf bh = loadB16(wrh, k, ksB);
    v16bf bl = loadB16(wrl, k, ksB);
    c = wmma_bf16(ah, bh, c);
    c = wmma_bf16(ah, bl, c);
    c = wmma_bf16(al, bh, c);
  }
  if (hrow < Hh) {
    int rbase = tm * 16 + ((lane >> 4) << 3);
#pragma unroll
    for (int r = 0; r < 8; ++r) C[(size_t)(rbase + r) * HPL + hrow] = c[r];
  }
}

// ---------------------------------------------------------------------------
// attn_h[t,b,d] = tanh( [c | input](b,t,:1024) . W_out[d,:1024] )  (bf16x3)
// ---------------------------------------------------------------------------
__global__ void k_attn_bf3(const u16* __restrict__ cH, const u16* __restrict__ cL,
                           const u16* __restrict__ inH, const u16* __restrict__ inL,
                           const u16* __restrict__ woH, const u16* __restrict__ woL,
                           float* __restrict__ out_attn) {
  int lane = threadIdx.x & 31;
  int wave = blockIdx.x * (blockDim.x >> 5) + (threadIdx.x >> 5);
  if (wave >= Bb * (Tt / 16) * (Dd / 16)) return;
  int bi = wave >> 8;
  int rt = wave & 255;
  int tm = rt >> 5;
  int tn = rt & 31;
  int r15 = lane & 15;
  int ksA = (lane >> 4) << 3;
  int ksB = (lane >> 4) << 4;
  size_t rowi = (size_t)(bi * Tt + tm * 16 + r15) * Dd;
  const u16* ach = cH + rowi;
  const u16* acl = cL + rowi;
  const u16* aih = inH + rowi;
  const u16* ail = inL + rowi;
  const u16* wrh = woH + (size_t)(tn * 16 + r15) * (2 * Dd);
  const u16* wrl = woL + (size_t)(tn * 16 + r15) * (2 * Dd);
  v8f c = {0.f, 0.f, 0.f, 0.f, 0.f, 0.f, 0.f, 0.f};
#pragma unroll 2
  for (int k = 0; k < 2 * Dd; k += 32) {  // 32-steps never straddle the 512 split
    v16bf ah, al;
    if (k < Dd) { ah = loadA16(ach, k, ksA);      al = loadA16(acl, k, ksA); }
    else        { ah = loadA16(aih, k - Dd, ksA); al = loadA16(ail, k - Dd, ksA); }
    v16bf bh = loadB16(wrh, k, ksB);
    v16bf bl = loadB16(wrl, k, ksB);
    c = wmma_bf16(ah, bh, c);
    c = wmma_bf16(ah, bl, c);
    c = wmma_bf16(al, bh, c);
  }
  int d = tn * 16 + r15;
  int tbase = tm * 16 + ((lane >> 4) << 3);
#pragma unroll
  for (int r = 0; r < 8; ++r)
    out_attn[(size_t)(tbase + r) * (Bb * Dd) + (size_t)bi * Dd + d] = tanhf(c[r]);
}

// ---------------------------------------------------------------------------
// Dominant fused kernel (bf16x3). One wave per (b,t,s0..s0+15).
// B rows clamped for g>=100; those columns yield finite garbage that is
// multiplied by wm=wv=0 in the epilogue.
// ---------------------------------------------------------------------------
__global__ void k_fused_h2(const float* __restrict__ ptb, const float* __restrict__ psb,
                           const float* __restrict__ b1,
                           const u16* __restrict__ w2h, const u16* __restrict__ w2l,
                           const float* __restrict__ b2,
                           const float* __restrict__ w_mean, const float* __restrict__ b_mean,
                           const float* __restrict__ w_var, const float* __restrict__ b_var,
                           float* __restrict__ out_mean, float* __restrict__ out_var) {
  __shared__ u16 hiS[4][16 * LSTR];
  __shared__ u16 loS[4][16 * LSTR];
  int lane  = threadIdx.x & 31;
  int wslot = threadIdx.x >> 5;
  int wave  = blockIdx.x * 4 + wslot;
  int b   = wave >> 10;
  int rem = wave & 1023;
  int t   = rem >> 3;
  int s0  = (rem & 7) << 4;
  u16* hi = hiS[wslot];
  u16* lo = loS[wslot];
  const float* ptrow  = ptb + (size_t)(b * Tt + t) * HPL;
  const float* psrows = psb + (size_t)(b * Ss + s0) * HPL;

  for (int idx = lane; idx < 16 * KP; idx += 32) {
    int m = idx >> 7;
    int k = idx & (KP - 1);
    float f = 0.f;
    if (k < Hh) f = softplusf(ptrow[k] + psrows[m * HPL + k] + b1[k]);
    u16 h = f2bf(f);
    hi[m * LSTR + k] = h;
    lo[m * LSTR + k] = f2bf(f - bf2f(h));
  }
  __syncthreads();

  int r15 = lane & 15;
  int ksA = (lane >> 4) << 3;
  int ksB = (lane >> 4) << 4;
  const u16* arh = hi + (size_t)r15 * LSTR;
  const u16* arl = lo + (size_t)r15 * LSTR;
  float accm[8] = {0.f, 0.f, 0.f, 0.f, 0.f, 0.f, 0.f, 0.f};
  float accv[8] = {0.f, 0.f, 0.f, 0.f, 0.f, 0.f, 0.f, 0.f};
  for (int nt = 0; nt < 7; ++nt) {
    int g = nt * 16 + r15;
    bool gok = (g < Hh);
    const u16* wrh = w2h + (size_t)(gok ? g : 0) * KP;
    const u16* wrl = w2l + (size_t)(gok ? g : 0) * KP;
    v8f c = {0.f, 0.f, 0.f, 0.f, 0.f, 0.f, 0.f, 0.f};
#pragma unroll
    for (int k = 0; k < KP; k += 32) {
      v16bf ah = loadA16(arh, k, ksA);
      v16bf al = loadA16(arl, k, ksA);
      v16bf bh = loadB16(wrh, k, ksB);
      v16bf bl = loadB16(wrl, k, ksB);
      c = wmma_bf16(ah, bh, c);
      c = wmma_bf16(ah, bl, c);
      c = wmma_bf16(al, bh, c);
    }
    float wm  = gok ? w_mean[g] : 0.f;
    float wv  = gok ? w_var[g]  : 0.f;
    float bb2 = gok ? b2[g]     : 0.f;
#pragma unroll
    for (int r = 0; r < 8; ++r) {
      float h2v = softplusf(c[r] + bb2);
      accm[r] += h2v * wm;
      accv[r] += h2v * wv;
    }
  }
#pragma unroll
  for (int m = 8; m >= 1; m >>= 1) {
#pragma unroll
    for (int r = 0; r < 8; ++r) {
      accm[r] += __shfl_xor(accm[r], m, 32);
      accv[r] += __shfl_xor(accv[r], m, 32);
    }
  }
  if (r15 == 0) {
    float bm = b_mean[0], bv = b_var[0];
    int srowi = s0 + ((lane >> 4) << 3);
    size_t base = (size_t)(b * Tt + t) * Ss + srowi;
#pragma unroll
    for (int r = 0; r < 8; ++r) {
      out_mean[base + r] = softplusf(accm[r] + bm);
      out_var [base + r] = softplusf(accv[r] + bv);
    }
  }
}

}  // namespace

extern "C" void kernel_launch(void* const* d_in, const int* in_sizes, int n_in,
                              void* d_out, int out_size, void* d_ws, size_t ws_size,
                              hipStream_t stream) {
  (void)in_sizes; (void)n_in; (void)out_size; (void)ws_size;
  const float* input  = (const float*)d_in[0];
  const float* mem    = (const float*)d_in[1];
  const float* W_in   = (const float*)d_in[2];
  const float* W1     = (const float*)d_in[3];
  const float* b1     = (const float*)d_in[4];
  const float* W2     = (const float*)d_in[5];
  const float* b2     = (const float*)d_in[6];
  const float* w_mean = (const float*)d_in[7];
  const float* b_mean = (const float*)d_in[8];
  const float* w_var  = (const float*)d_in[9];
  const float* b_var  = (const float*)d_in[10];
  const float* W_out  = (const float*)d_in[11];

  float* out_attn  = (float*)d_out;                    // (T,B,D)
  float* out_align = out_attn + (size_t)Tt * Bb * Dd;  // (T,B,S)
  float* out_mean  = out_align + (size_t)Tt * Bb * Ss; // (B,T,S)
  float* out_var   = out_mean + (size_t)Bb * Tt * Ss;  // (B,T,S)

  const int nIn  = Bb * Tt * Dd;   // 1048576
  const int nMem = Bb * Ss * Dd;   // 1048576
  const int nWin = Dd * Dd;        // 262144
  const int nW1  = Hh * 2 * Dd;    // 102400
  const int nWo  = Dd * 2 * Dd;    // 524288
  const int nAl  = Bb * Tt * Ss;   // 262144

  float* ws       = (float*)d_ws;
  float* ht       = ws;                       // nIn f32
  float* alignbuf = ht + (size_t)nIn;         // nAl f32
  float* cbuf     = alignbuf + (size_t)nAl;   // nIn f32
  float* ptb      = cbuf + (size_t)nIn;       // 2048*HPL f32
  float* psb      = ptb + (size_t)Bb * Tt * HPL;
  u16* u = (u16*)(psb + (size_t)Bb * Ss * HPL);
  u16* w2h = u;               u += (size_t)Hh * KP;
  u16* w2l = u;               u += (size_t)Hh * KP;
  u16* inH = u;               u += nIn;
  u16* inL = u;               u += nIn;
  u16* memH = u;              u += nMem;
  u16* memL = u;              u += nMem;
  u16* memTH = u;             u += nMem;
  u16* memTL = u;             u += nMem;
  u16* winH = u;              u += nWin;
  u16* winL = u;              u += nWin;
  u16* w1H = u;               u += nW1;
  u16* w1L = u;               u += nW1;
  u16* woH = u;               u += nWo;
  u16* woL = u;               u += nWo;
  u16* htH = u;               u += nIn;
  u16* htL = u;               u += nIn;
  u16* avH = u;               u += nAl;
  u16* avL = u;               u += nAl;
  u16* cH = u;                u += nIn;
  u16* cL = u;                /* total ws ~= 37 MB */

  const dim3 blk(256);
  auto nb = [](int n) { return dim3((unsigned)((n + 255) / 256)); };

  // --- operand splits (hi/lo bf16) ---
  k_split<<<nb(nIn),  blk, 0, stream>>>(input, inH, inL, nIn);
  k_split<<<nb(nMem), blk, 0, stream>>>(mem, memH, memL, nMem);
  k_split<<<nb(nWin), blk, 0, stream>>>(W_in, winH, winL, nWin);
  k_split<<<nb(nW1),  blk, 0, stream>>>(W1, w1H, w1L, nW1);
  k_split<<<nb(nWo),  blk, 0, stream>>>(W_out, woH, woL, nWo);
  k_tsplit_mem<<<nb(nMem), blk, 0, stream>>>(mem, memTH, memTL);
  k_prep_w2<<<nb(Hh * KP), blk, 0, stream>>>(W2, w2h, w2l);

  // 1) ht = input @ W_in^T : M=2048 N=512 K=512
  k_gemm_bf3<<<dim3(512), blk, 0, stream>>>(inH, inL, winH, winL, ht,
      128, 32, Dd, Dd, Dd, Dd, 0, 0, 0, 1);

  // 2) pt / ps projections (bf16x3, store-guarded N=100)
  k_pt_ps_bf3<<<dim3(224), blk, 0, stream>>>(inH, inL, memH, memL, w1H, w1L, ptb, psb);

  // 3) align[b] = ht[b] @ mem[b]^T : per-b M=T N=S K=D
  k_split<<<nb(nIn), blk, 0, stream>>>(ht, htH, htL, nIn);
  k_gemm_bf3<<<dim3(128), blk, 0, stream>>>(htH, htL, memH, memL, alignbuf,
      8, 8, Dd, Dd, Dd, Ss, (long)Tt * Dd, (long)Ss * Dd, (long)Tt * Ss, Bb);

  // 4) softmax + (T,B,S) copy into d_out, then split av
  k_softmax<<<dim3(256), blk, 0, stream>>>(alignbuf, out_align);
  k_split<<<nb(nAl), blk, 0, stream>>>(alignbuf, avH, avL, nAl);

  // 5) c[b] = av[b] @ mem[b] == av[b] @ memT[b]^T : per-b M=T N=D K=S
  k_gemm_bf3<<<dim3(512), blk, 0, stream>>>(avH, avL, memTH, memTL, cbuf,
      8, 32, Ss, Ss, Ss, Dd, (long)Tt * Ss, (long)Dd * Ss, (long)Tt * Dd, Bb);

  // 6) attn_h = tanh([c|input] @ W_out^T), transposed store
  k_split<<<nb(nIn), blk, 0, stream>>>(cbuf, cH, cL, nIn);
  k_attn_bf3<<<dim3(512), blk, 0, stream>>>(cH, cL, inH, inL, woH, woL, out_attn);

  // 7) fused h1->h2->mean/var (16384 wave tiles, 4 waves/block)
  k_fused_h2<<<dim3(4096), dim3(128), 0, stream>>>(ptb, psb, b1, w2h, w2l, b2,
      w_mean, b_mean, w_var, b_var, out_mean, out_var);
}
